// Encoder_Adversarial_GraphSAGE_21904333210050
// MI455X (gfx1250) — matrix operand order repaired and checked by reading, not compiled, not run on hardware
//
#include <hip/hip_runtime.h>

typedef __attribute__((ext_vector_type(16))) __bf16 v16bf;
typedef __attribute__((ext_vector_type(8)))  float  v8f;

// ---------------------------------------------------------------------------
// Native CDNA5 fire-and-forget FP32 global atomic add (no return -> STOREcnt).
// Device scope so cross-workgroup accumulation is coherent in L2's atomic units.
// ---------------------------------------------------------------------------
__device__ __forceinline__ void gatomic_add_f32(float* p, float v) {
  asm volatile("global_atomic_add_f32 %0, %1, off scope:SCOPE_DEV"
               :: "v"(p), "v"(v) : "memory");
}

// ---------------------------------------------------------------------------
// Utility: zero a float4-aligned region
// ---------------------------------------------------------------------------
__global__ void zero4_kernel(float4* __restrict__ p, long n4) {
  long i = (long)blockIdx.x * blockDim.x + threadIdx.x;
  if (i < n4) p[i] = make_float4(0.f, 0.f, 0.f, 0.f);
}

// ---------------------------------------------------------------------------
// Degree count + inverse (mean normalization folded into GEMM later)
// ---------------------------------------------------------------------------
__global__ void degree_kernel(const int* __restrict__ dst, float* __restrict__ deg, int E) {
  int e = blockIdx.x * blockDim.x + threadIdx.x;
  if (e < E) gatomic_add_f32(&deg[dst[e]], 1.0f);
}

__global__ void invdeg_kernel(float* __restrict__ deg, int n) {
  int i = blockIdx.x * blockDim.x + threadIdx.x;
  if (i < n) deg[i] = 1.0f / fmaxf(deg[i], 1.0f);
}

// ---------------------------------------------------------------------------
// Edge scatter-add: one wave (32 lanes) per edge; each lane owns D/32 floats.
// agg[dst] += X[src]  (gathers are L2-resident: 51-102MB rows vs 192MB L2)
// ---------------------------------------------------------------------------
template<int D>
__global__ void scatter_add_kernel(const float* __restrict__ X, const int* __restrict__ src,
                                   const int* __restrict__ dst, float* __restrict__ Agg, int E) {
  long gid = (long)blockIdx.x * blockDim.x + threadIdx.x;
  int e = (int)(gid >> 5);
  if (e >= E) return;
  int lane = (int)(gid & 31);
  const float* xs = X + (size_t)src[e] * D;
  float*       ad = Agg + (size_t)dst[e] * D;
  constexpr int C = D / 32;
#pragma unroll
  for (int j = 0; j < C; j += 4) {
    int off = lane * C + j;
    float4 v = *(const float4*)(xs + off);
    gatomic_add_f32(ad + off + 0, v.x);
    gatomic_add_f32(ad + off + 1, v.y);
    gatomic_add_f32(ad + off + 2, v.z);
    gatomic_add_f32(ad + off + 3, v.w);
  }
}

// ---------------------------------------------------------------------------
// Pre-swizzle f32 weight matrix [K, NC] (row-major) into bf16 WMMA B-fragments.
// B fragment k order used here (contiguous, 16x16x32 bf16, wave32): lane = {half = lane>>4, n = lane&15},
//   VGPR j holds K = half*16 + 2j (lo) and +1 (hi).
// Fragment storage: [ (s0*NT + t) * 32 + lane ] * 16 contiguous bf16.
// ---------------------------------------------------------------------------
template<int K, int NC>
__global__ void prepack_kernel(const float* __restrict__ W, __bf16* __restrict__ Wp) {
  constexpr int NT = NC / 16;
  constexpr int KS = K / 32;
  int gid = blockIdx.x * blockDim.x + threadIdx.x;
  if (gid >= KS * NT * 32) return;
  int lane = gid & 31;
  int frag = gid >> 5;
  int t  = frag % NT;
  int s0 = frag / NT;
  int half = lane >> 4, ln = lane & 15;
  int n = t * 16 + ln;
  __bf16* o = Wp + (size_t)gid * 16;
#pragma unroll
  for (int j = 0; j < 8; ++j) {
    int k = s0 * 32 + half * 16 + 2 * j;
    o[2 * j]     = (__bf16)W[(size_t)k * NC + n];
    o[2 * j + 1] = (__bf16)W[(size_t)(k + 1) * NC + n];
  }
}

// ---------------------------------------------------------------------------
// Fused SAGE GEMM:  Out = (Agg * inv[row]) @ WL + X @ WR + bias + perturb
// Block = 256 threads (8 waves), computes 16 rows x NC cols.
// A fragment (16x32 bf16): lane m = lane&15,
//   VGPR j: K = (lane>>4)*16 + 2j,+1  (matches prepacked B k order)
// C/D: VGPR r -> row m = r + 8*(lane>>4), col n = lane&15.
// ---------------------------------------------------------------------------
template<int K, int NC>
__global__ void __launch_bounds__(256)
sage_gemm_kernel(const float* __restrict__ Agg, const float* __restrict__ X,
                 const float* __restrict__ inv,
                 const __bf16* __restrict__ WpL, const __bf16* __restrict__ WpR,
                 const float* __restrict__ bias, const float* __restrict__ perturb,
                 float* __restrict__ Out) {
  constexpr int NT = NC / 16;   // column tiles
  constexpr int KS = K / 32;    // k steps
  constexpr int TP = NT / 8;    // tiles per wave (8 waves)
  int w    = threadIdx.x >> 5;
  int lane = threadIdx.x & 31;
  int half = lane >> 4, ln = lane & 15;
  int rowA = blockIdx.x * 16 + ln;
  float rscale = inv[rowA];
  int kb = half * 8;

  v8f acc[TP] = {};

#pragma unroll
  for (int pass = 0; pass < 2; ++pass) {
    const float*  arow = (pass ? X : Agg) + (size_t)rowA * K;
    const __bf16* Wp   = pass ? WpR : WpL;
    float s = pass ? 1.0f : rscale;
#pragma unroll
    for (int s0 = 0; s0 < KS; ++s0) {
      v16bf a;
#pragma unroll
      for (int j = 0; j < 8; ++j) {
        int kA = s0 * 32 + half * 16 + 2 * j;
        float2 f = *(const float2*)(arow + kA);
        a[2 * j]     = (__bf16)(f.x * s);
        a[2 * j + 1] = (__bf16)(f.y * s);
      }
#pragma unroll
      for (int tt = 0; tt < TP; ++tt) {
        int t = w * TP + tt;
        v16bf b = *(const v16bf*)(Wp + ((size_t)(s0 * NT + t) * 32 + lane) * 16);
        acc[tt] = __builtin_amdgcn_wmma_f32_16x16x32_bf16(
            false, a, false, b, (short)0, acc[tt], false, false);
      }
    }
  }

#pragma unroll
  for (int tt = 0; tt < TP; ++tt) {
    int n = (w * TP + tt) * 16 + ln;
    float bn = bias[n];
#pragma unroll
    for (int r = 0; r < 8; ++r) {
      int row = blockIdx.x * 16 + r + 8 * half;
      size_t idx = (size_t)row * NC + n;
      Out[idx] = acc[tt][r] + bn + perturb[idx];
    }
  }
}

// ---------------------------------------------------------------------------
// Driver
// ---------------------------------------------------------------------------
extern "C" void kernel_launch(void* const* d_in, const int* in_sizes, int n_in,
                              void* d_out, int out_size, void* d_ws, size_t ws_size,
                              hipStream_t stream) {
  const float* x   = (const float*)d_in[0];
  const int*   ei  = (const int*)d_in[1];   // JAX default x64-disabled -> int32
  const float* pf  = (const float*)d_in[2];
  const float* pl  = (const float*)d_in[3];
  const float* Wl1 = (const float*)d_in[4];
  const float* bl1 = (const float*)d_in[5];
  const float* Wr1 = (const float*)d_in[6];
  const float* Wl2 = (const float*)d_in[7];
  const float* bl2 = (const float*)d_in[8];
  const float* Wr2 = (const float*)d_in[9];
  float* out = (float*)d_out;

  const int N = in_sizes[0] / 128;          // 100000
  const int E = in_sizes[1] / 2;            // 1600000
  const int* src = ei;
  const int* dst = ei + E;

  float* ws   = (float*)d_ws;
  float* deg  = ws;                              // N          (becomes inv-deg)
  float* agg1 = deg + N;                         // N*128
  float* h    = agg1 + (size_t)N * 128;          // N*256
  float* agg2 = h + (size_t)N * 256;             // N*256
  __bf16* wp1l = (__bf16*)(agg2 + (size_t)N * 256);  // 4*16*32*16 = 32768 each
  __bf16* wp1r = wp1l + 32768;
  __bf16* wp2l = wp1r + 32768;
  __bf16* wp2r = wp2l + 32768;

  const int T = 256;

  // zero deg + agg1 (contiguous region, N*(1+128) floats, multiple of 4)
  {
    long n4 = (long)N * 129 / 4;
    zero4_kernel<<<(unsigned)((n4 + T - 1) / T), T, 0, stream>>>((float4*)deg, n4);
  }

  // pre-swizzle weights to bf16 WMMA fragments (tiny, once per launch)
  prepack_kernel<128, 256><<<(4 * 16 * 32 + T - 1) / T, T, 0, stream>>>(Wl1, wp1l);
  prepack_kernel<128, 256><<<(4 * 16 * 32 + T - 1) / T, T, 0, stream>>>(Wr1, wp1r);
  prepack_kernel<256, 128><<<(8 * 8 * 32 + T - 1) / T, T, 0, stream>>>(Wl2, wp2l);
  prepack_kernel<256, 128><<<(8 * 8 * 32 + T - 1) / T, T, 0, stream>>>(Wr2, wp2r);

  // degrees (shared by both layers), then invert in place
  degree_kernel<<<(E + T - 1) / T, T, 0, stream>>>(dst, deg, E);
  invdeg_kernel<<<(N + T - 1) / T, T, 0, stream>>>(deg, N);

  // layer 1: scatter x -> agg1, then fused GEMM -> h
  {
    long total = (long)E * 32;
    scatter_add_kernel<128><<<(unsigned)((total + T - 1) / T), T, 0, stream>>>(x, src, dst, agg1, E);
  }
  sage_gemm_kernel<128, 256><<<N / 16, T, 0, stream>>>(agg1, x, deg, wp1l, wp1r, bl1, pf, h);

  // layer 2: zero agg2, scatter h -> agg2, fused GEMM -> out
  {
    long n4 = (long)N * 256 / 4;
    zero4_kernel<<<(unsigned)((n4 + T - 1) / T), T, 0, stream>>>((float4*)agg2, n4);
  }
  {
    long total = (long)E * 32;
    scatter_add_kernel<256><<<(unsigned)((total + T - 1) / T), T, 0, stream>>>(h, src, dst, agg2, E);
  }
  sage_gemm_kernel<256, 128><<<N / 16, T, 0, stream>>>(agg2, h, deg, wp2l, wp2r, bl2, pl, out);

  (void)n_in; (void)out_size; (void)ws_size;
}
